// Transformer_60224031424668
// MI455X (gfx1250) — compile-verified, matrix operand order checked by reading, not run on hardware
//
#include <hip/hip_runtime.h>
#include <hip/hip_bf16.h>

// ---------------------------------------------------------------------------
// Transformer forward for MI455X (gfx1250), all dense math via WMMA f16.
// ---------------------------------------------------------------------------

typedef __attribute__((ext_vector_type(16))) _Float16 v16h;
typedef __attribute__((ext_vector_type(8)))  _Float16 v8h;
typedef __attribute__((ext_vector_type(4)))  _Float16 v4h;
typedef __attribute__((ext_vector_type(8)))  float    v8f;

// Problem constants (match reference)
#define NL   6
#define DIM  512
#define NH   8
#define HD   64
#define FFD  2048
#define VOC  32000
#define NB   8
#define SEQ  512
#define ROWS (NB * SEQ)   // 4096

// GEMM tiling: block = 256 threads = 8 waves (wave32).
// Waves arranged 4 (rows) x 2 (cols); each wave computes 32x32 via 2x2 WMMA tiles.
// Block tile: 128 (M) x 64 (N), K step 32.
#define BM 128
#define BN 64
#define BK 32
#define LDS_STRIDE 40   // halfwords per row; multiple of 8 -> 16B aligned runs

// ---------------------------------------------------------------------------
// WMMA GEMM:  C[m,n] = act( sum_k A[m,k]*B[k,n] + bias[n] + resid[m,n] )
// TRANSB==true means B is supplied as [N,K] row-major (used for Q @ K^T).
// Batched via blockIdx.z: base = (z/Hdiv)*outer + (z%Hdiv)*inner per operand.
// All dims must be multiples of the tile sizes (true for this model).
// ---------------------------------------------------------------------------
template <bool TRANSB, bool RELU>
__global__ __launch_bounds__(256)
void wmma_gemm(const float* __restrict__ A,  int lda, long long aOuter, long long aInner,
               const float* __restrict__ Bm, int ldb, long long bOuter, long long bInner,
               const float* __restrict__ bias,
               const float* __restrict__ resid,
               float* __restrict__ C,  int ldc, long long cOuter, long long cInner,
               int Hdiv, int K)
{
    __shared__ alignas(16) _Float16 As[BM * LDS_STRIDE];
    __shared__ alignas(16) _Float16 Bs[BN * LDS_STRIDE];  // stored as B^T: [n][k]

    const int tid  = threadIdx.x;
    const int lane = tid & 31;
    const int wave = tid >> 5;
    const int wRow = wave >> 1;          // 0..3
    const int wCol = wave & 1;           // 0..1
    const int l15  = lane & 15;
    const int kb   = (lane >> 4) << 3;   // 0 or 8 (K sub-block base per ISA layout)

    const int gm0 = blockIdx.y * BM;
    const int gn0 = blockIdx.x * BN;

    const int z  = blockIdx.z;
    const int bb = z / Hdiv;
    const int hh = z % Hdiv;
    const long long aBase = (long long)bb * aOuter + (long long)hh * aInner;
    const long long bBase = (long long)bb * bOuter + (long long)hh * bInner;
    const long long cBase = (long long)bb * cOuter + (long long)hh * cInner;

    v8f acc[2][2] = {};

    #pragma unroll 2
    for (int k0 = 0; k0 < K; k0 += BK) {
        // ---- stage A tile (BM x BK floats = 1024 float4), fp32 -> f16 ----
        // Constant trip count so the compiler fully unrolls (no exec-mask loop);
        // float4 loads -> global_load_b128, packed v4h store -> ds_store_b64.
        // Unconditional speculative prefetch (locality 3 -> WGP scope) of the
        // next K tile; harmless past the end (dropped on translation miss).
        #pragma unroll
        for (int i = 0; i < 4; ++i) {
            const int f  = tid + i * 256;          // 0..1023
            const int r  = f >> 3;                 // 0..127
            const int c4 = (f & 7) << 2;           // 0,4,...,28
            const float* gp = &A[aBase + (long long)(gm0 + r) * lda + (k0 + c4)];
            const float4 v = *reinterpret_cast<const float4*>(gp);
            __builtin_prefetch(gp + BK, 0, 3);     // next K tile, WGP scope
            v4h h;
            h[0] = (_Float16)v.x; h[1] = (_Float16)v.y;
            h[2] = (_Float16)v.z; h[3] = (_Float16)v.w;
            *reinterpret_cast<v4h*>(&As[r * LDS_STRIDE + c4]) = h;
        }

        // ---- stage B tile as B^T (BN x BK floats = 512 float4) ----
        if (TRANSB) {
            #pragma unroll
            for (int i = 0; i < 2; ++i) {
                const int f  = tid + i * 256;      // 0..511
                const int n  = f >> 3;             // 0..63
                const int c4 = (f & 7) << 2;       // 0,4,...,28
                const float* gp = &Bm[bBase + (long long)(gn0 + n) * ldb + (k0 + c4)];
                const float4 v = *reinterpret_cast<const float4*>(gp);
                __builtin_prefetch(gp + BK, 0, 3);
                v4h h;
                h[0] = (_Float16)v.x; h[1] = (_Float16)v.y;
                h[2] = (_Float16)v.z; h[3] = (_Float16)v.w;
                *reinterpret_cast<v4h*>(&Bs[n * LDS_STRIDE + c4]) = h;
            }
        } else {
            #pragma unroll
            for (int i = 0; i < 2; ++i) {
                const int f  = tid + i * 256;      // 0..511
                const int n4 = (f & 15) << 2;      // 0,4,...,60
                const int c  = f >> 4;             // 0..31
                const float* gp = &Bm[bBase + (long long)(k0 + c) * ldb + (gn0 + n4)];
                const float4 v = *reinterpret_cast<const float4*>(gp);
                __builtin_prefetch(gp + (long long)BK * ldb, 0, 3);  // next K tile
                Bs[(n4 + 0) * LDS_STRIDE + c] = (_Float16)v.x;
                Bs[(n4 + 1) * LDS_STRIDE + c] = (_Float16)v.y;
                Bs[(n4 + 2) * LDS_STRIDE + c] = (_Float16)v.z;
                Bs[(n4 + 3) * LDS_STRIDE + c] = (_Float16)v.w;
            }
        }
        __syncthreads();

        // ---- build fragments per CDNA5 16-bit A/B VGPR layout ----
        // lanes 0-15:  elements 0..7 -> K 0..7,  8..15 -> K 16..23
        // lanes 16-31: elements 0..7 -> K 8..15, 8..15 -> K 24..31
        v16h af[2], bf[2];
        #pragma unroll
        for (int mi = 0; mi < 2; ++mi) {
            const _Float16* p = &As[(wRow * 32 + mi * 16 + l15) * LDS_STRIDE + kb];
            v8h lo = *reinterpret_cast<const v8h*>(p);
            v8h hi = *reinterpret_cast<const v8h*>(p + 16);
            #pragma unroll
            for (int e = 0; e < 8; ++e) { af[mi][e] = lo[e]; af[mi][e + 8] = hi[e]; }
        }
        #pragma unroll
        for (int ni = 0; ni < 2; ++ni) {
            const _Float16* p = &Bs[(wCol * 32 + ni * 16 + l15) * LDS_STRIDE + kb];
            v8h lo = *reinterpret_cast<const v8h*>(p);
            v8h hi = *reinterpret_cast<const v8h*>(p + 16);
            #pragma unroll
            for (int e = 0; e < 8; ++e) { bf[ni][e] = lo[e]; bf[ni][e + 8] = hi[e]; }
        }

        #pragma unroll
        for (int mi = 0; mi < 2; ++mi)
            #pragma unroll
            for (int ni = 0; ni < 2; ++ni)
                acc[mi][ni] = __builtin_amdgcn_wmma_f32_16x16x32_f16(
                    false, af[mi], false, bf[ni], (short)0, acc[mi][ni], false, false);

        __syncthreads();
    }

    // ---- epilogue: C/D layout -> lane n = lane&15, rows r + 8*(lane>=16) ----
    #pragma unroll
    for (int mi = 0; mi < 2; ++mi) {
        #pragma unroll
        for (int ni = 0; ni < 2; ++ni) {
            const int gn = gn0 + wCol * 32 + ni * 16 + l15;
            const int gm = gm0 + wRow * 32 + mi * 16 + ((lane >> 4) << 3);
            const float bv = bias ? bias[gn] : 0.0f;
            #pragma unroll
            for (int r = 0; r < 8; ++r) {
                long long off = cBase + (long long)(gm + r) * ldc + gn;
                float val = acc[mi][ni][r] + bv;
                if (resid) val += resid[off];
                if (RELU)  val = fmaxf(val, 0.0f);
                C[off] = val;
            }
        }
    }
}

// ---------------------------------------------------------------------------
// Embedding + sinusoidal positional encoding:  out[r,d] = emb[id[r],d]*sqrt(D)+pe
// ---------------------------------------------------------------------------
__global__ __launch_bounds__(256)
void embed_kernel(const int* __restrict__ ids, const float* __restrict__ emb,
                  float* __restrict__ out)
{
    const int rs  = blockIdx.x;          // b*SEQ + s
    const int pos = rs & (SEQ - 1);
    const int tok = ids[rs];
    const float SQ = 22.62741699796952f; // sqrt(512)
    #pragma unroll
    for (int i = 0; i < 2; ++i) {
        const int d = threadIdx.x + i * 256;
        int   de  = d & ~1;
        float div = __expf(-__logf(10000.0f) * (float)de / (float)DIM);
        float ang = (float)pos * div;
        float pe  = (d & 1) ? __cosf(ang) : __sinf(ang);
        out[(long long)rs * DIM + d] = emb[(long long)tok * DIM + d] * SQ + pe;
    }
}

// ---------------------------------------------------------------------------
// LayerNorm over last dim (512), one block per row.
// ---------------------------------------------------------------------------
__global__ __launch_bounds__(256)
void layernorm_kernel(const float* __restrict__ x, const float* __restrict__ g,
                      const float* __restrict__ b, float* __restrict__ out)
{
    __shared__ float red[256];
    const int row = blockIdx.x, tid = threadIdx.x;
    const float* xr = x + (long long)row * DIM;
    float v0 = xr[tid], v1 = xr[tid + 256];

    red[tid] = v0 + v1;
    __syncthreads();
    for (int s = 128; s > 0; s >>= 1) { if (tid < s) red[tid] += red[tid + s]; __syncthreads(); }
    const float mean = red[0] * (1.0f / DIM);
    __syncthreads();

    float d0 = v0 - mean, d1 = v1 - mean;
    red[tid] = d0 * d0 + d1 * d1;
    __syncthreads();
    for (int s = 128; s > 0; s >>= 1) { if (tid < s) red[tid] += red[tid + s]; __syncthreads(); }
    const float inv = rsqrtf(red[0] * (1.0f / DIM) + 1e-5f);

    float* o = out + (long long)row * DIM;
    o[tid]       = d0 * inv * g[tid]       + b[tid];
    o[tid + 256] = d1 * inv * g[tid + 256] + b[tid + 256];
}

// ---------------------------------------------------------------------------
// Masked, scaled softmax (in place) over last dim of scores[z][q][k].
// mask: token ids[b*SEQ+k]==0 (padding) OR (causal && k>q). scale = 1/sqrt(64).
// ---------------------------------------------------------------------------
__global__ __launch_bounds__(256)
void softmax_kernel(float* __restrict__ scores, const int* __restrict__ ids, int causal)
{
    __shared__ float red[256];
    const int q   = blockIdx.x;
    const int z   = blockIdx.y;          // b*NH + h
    const int tid = threadIdx.x;
    const int bb  = z >> 3;              // / NH
    float* row = scores + ((long long)z * SEQ + q) * SEQ;

    const int k0 = tid, k1 = tid + 256;
    const bool m0 = (ids[bb * SEQ + k0] == 0) || (causal && k0 > q);
    const bool m1 = (ids[bb * SEQ + k1] == 0) || (causal && k1 > q);
    float s0 = m0 ? -1e9f : row[k0] * 0.125f;
    float s1 = m1 ? -1e9f : row[k1] * 0.125f;

    red[tid] = fmaxf(s0, s1);
    __syncthreads();
    for (int s = 128; s > 0; s >>= 1) { if (tid < s) red[tid] = fmaxf(red[tid], red[tid + s]); __syncthreads(); }
    const float mx = red[0];
    __syncthreads();

    float e0 = __expf(s0 - mx), e1 = __expf(s1 - mx);
    red[tid] = e0 + e1;
    __syncthreads();
    for (int s = 128; s > 0; s >>= 1) { if (tid < s) red[tid] += red[tid + s]; __syncthreads(); }
    const float rs = 1.0f / red[0];

    row[k0] = e0 * rs;
    row[k1] = e1 * rs;
}

__global__ __launch_bounds__(256)
void add_kernel(const float* __restrict__ a, const float* __restrict__ b,
                float* __restrict__ c, long long n)
{
    long long i = (long long)blockIdx.x * 256 + threadIdx.x;
    if (i < n) c[i] = a[i] + b[i];
}

// ---------------------------------------------------------------------------
// Host side
// ---------------------------------------------------------------------------
static void gemm(hipStream_t st, bool transB, bool relu,
                 const float* A, int lda, long long aO, long long aI,
                 const float* Bm, int ldb, long long bO, long long bI,
                 const float* bias, const float* resid,
                 float* C, int ldc, long long cO, long long cI,
                 int Hdiv, int M, int N, int K, int batches)
{
    dim3 g(N / BN, M / BM, batches), b(256);
    if (transB)
        wmma_gemm<true,  false><<<g, b, 0, st>>>(A, lda, aO, aI, Bm, ldb, bO, bI,
                                                 bias, resid, C, ldc, cO, cI, Hdiv, K);
    else if (relu)
        wmma_gemm<false, true ><<<g, b, 0, st>>>(A, lda, aO, aI, Bm, ldb, bO, bI,
                                                 bias, resid, C, ldc, cO, cI, Hdiv, K);
    else
        wmma_gemm<false, false><<<g, b, 0, st>>>(A, lda, aO, aI, Bm, ldb, bO, bI,
                                                 bias, resid, C, ldc, cO, cI, Hdiv, K);
}

// Plain (non-batched) dense layer: C[M,N] = act(A[M,K] @ W[K,N] + bias (+resid))
static void dense(hipStream_t st, const float* A, const float* W, const float* bias,
                  const float* resid, float* C, int M, int N, int K, bool relu)
{
    gemm(st, false, relu, A, K, 0, 0, W, N, 0, 0, bias, resid, C, N, 0, 0, 1, M, N, K, 1);
}

// Full multi-head attention block. out = ctx @ Wo + bo (+ resid).
static void attention(hipStream_t st,
                      const float* hq, const float* kvsrc,
                      const float* Wq, const float* bq, const float* Wk, const float* bk,
                      const float* Wv, const float* bv, const float* Wo, const float* bo,
                      const int* mask_ids, int causal, const float* resid, float* out,
                      float* Qb, float* Kb, float* Vb, float* Sc, float* CT)
{
    const long long SD = (long long)SEQ * DIM;
    const long long SS = (long long)SEQ * SEQ;
    dense(st, hq,    Wq, bq, nullptr, Qb, ROWS, DIM, DIM, false);
    dense(st, kvsrc, Wk, bk, nullptr, Kb, ROWS, DIM, DIM, false);
    dense(st, kvsrc, Wv, bv, nullptr, Vb, ROWS, DIM, DIM, false);
    // scores[z,q,k] = sum_d Q[b,q,h,d] * K[b,k,h,d]     (B^T layout is natural)
    gemm(st, true, false,
         Qb, DIM, SD, HD,
         Kb, DIM, SD, HD,
         nullptr, nullptr,
         Sc, SEQ, (long long)NH * SS, SS,
         NH, SEQ, SEQ, HD, NB * NH);
    softmax_kernel<<<dim3(SEQ, NB * NH), 256, 0, st>>>(Sc, mask_ids, causal);
    // ctx[b,q,h,d] = sum_k P[z,q,k] * V[b,k,h,d]
    gemm(st, false, false,
         Sc, SEQ, (long long)NH * SS, SS,
         Vb, DIM, SD, HD,
         nullptr, nullptr,
         CT, DIM, SD, HD,
         NH, SEQ, HD, SEQ, NB * NH);
    dense(st, CT, Wo, bo, resid, out, ROWS, DIM, DIM, false);
}

// Flat input order (depth-first over setup_inputs() dict insertion order)
enum {
    IN_ENC = 0, IN_DEC, ENC_EMB, DEC_EMB,
    E_WQ, E_BQ, E_WK, E_BK, E_WV, E_BV, E_WO, E_BO,     // 4..11  enc.attn
    E_W1, E_B1, E_W2, E_B2,                             // 12..15 enc.ffn
    E_LN1G, E_LN1B, E_LN2G, E_LN2B,                     // 16..19
    ENC_NG, ENC_NB,                                     // 20..21
    DS_WQ, DS_BQ, DS_WK, DS_BK, DS_WV, DS_BV, DS_WO, DS_BO,  // 22..29 dec.self
    DC_WQ, DC_BQ, DC_WK, DC_BK, DC_WV, DC_BV, DC_WO, DC_BO,  // 30..37 dec.cross
    D_W1, D_B1, D_W2, D_B2,                             // 38..41 dec.ffn
    D_LN1G, D_LN1B, D_LN2G, D_LN2B, D_LN3G, D_LN3B,     // 42..47
    FIN_W, FIN_B                                        // 48..49
};

extern "C" void kernel_launch(void* const* d_in, const int* in_sizes, int n_in,
                              void* d_out, int out_size, void* d_ws, size_t ws_size,
                              hipStream_t stream)
{
    (void)in_sizes; (void)n_in; (void)out_size; (void)ws_size;
    auto F = [&](int i) { return (const float*)d_in[i]; };
    const int* enc_ids = (const int*)d_in[IN_ENC];
    const int* dec_ids = (const int*)d_in[IN_DEC];

    const long long ND = (long long)ROWS * DIM;     // 2,097,152 floats
    const long long DD = (long long)DIM * DIM;
    const long long DF = (long long)DIM * FFD;

    float* ws = (float*)d_ws;
    float* X  = ws;                 // encoder state      [ROWS, DIM]
    float* Hb = ws + 1 * ND;        // LN output
    float* Qb = ws + 2 * ND;
    float* Kb = ws + 3 * ND;
    float* Vb = ws + 4 * ND;
    float* CT = ws + 5 * ND;        // attention context
    float* CR = ws + 6 * ND;        // decoder cross output
    float* Tb = ws + 7 * ND;        // y + cross
    float* Yb = ws + 8 * ND;        // decoder state
    float* Fb = ws + 9 * ND;        // FFN hidden [ROWS, FFD] (4*ND)
    float* Sc = ws + 13 * ND;       // scores [B*H, S, S]    (8*ND)

    float* outp   = (float*)d_out;
    float* encOut = outp;           // [ROWS, DIM]
    float* logits = outp + ND;      // [ROWS, VOC]

    // ---------------- encoder ----------------
    embed_kernel<<<ROWS, 256, 0, stream>>>(enc_ids, F(ENC_EMB), X);
    for (int i = 0; i < NL; ++i) {
        layernorm_kernel<<<ROWS, 256, 0, stream>>>(X, F(E_LN1G) + i * DIM, F(E_LN1B) + i * DIM, Hb);
        attention(stream, Hb, Hb,
                  F(E_WQ) + i * DD, F(E_BQ) + i * DIM, F(E_WK) + i * DD, F(E_BK) + i * DIM,
                  F(E_WV) + i * DD, F(E_BV) + i * DIM, F(E_WO) + i * DD, F(E_BO) + i * DIM,
                  enc_ids, 0, /*resid=*/X, /*out=*/X, Qb, Kb, Vb, Sc, CT);
        layernorm_kernel<<<ROWS, 256, 0, stream>>>(X, F(E_LN2G) + i * DIM, F(E_LN2B) + i * DIM, Hb);
        dense(stream, Hb, F(E_W1) + i * DF, F(E_B1) + i * FFD, nullptr, Fb, ROWS, FFD, DIM, true);
        dense(stream, Fb, F(E_W2) + i * DF, F(E_B2) + i * DIM, X, X, ROWS, DIM, FFD, false);
    }
    layernorm_kernel<<<ROWS, 256, 0, stream>>>(X, F(ENC_NG), F(ENC_NB), encOut);

    // ---------------- decoder ----------------
    embed_kernel<<<ROWS, 256, 0, stream>>>(dec_ids, F(DEC_EMB), Yb);
    for (int i = 0; i < NL; ++i) {
        // self-attention (causal + dec padding)
        layernorm_kernel<<<ROWS, 256, 0, stream>>>(Yb, F(D_LN1G) + i * DIM, F(D_LN1B) + i * DIM, Hb);
        attention(stream, Hb, Hb,
                  F(DS_WQ) + i * DD, F(DS_BQ) + i * DIM, F(DS_WK) + i * DD, F(DS_BK) + i * DIM,
                  F(DS_WV) + i * DD, F(DS_BV) + i * DIM, F(DS_WO) + i * DD, F(DS_BO) + i * DIM,
                  dec_ids, 1, /*resid=*/Yb, /*out=*/Yb, Qb, Kb, Vb, Sc, CT);
        // cross-attention (enc padding)
        layernorm_kernel<<<ROWS, 256, 0, stream>>>(Yb, F(D_LN2G) + i * DIM, F(D_LN2B) + i * DIM, Hb);
        attention(stream, Hb, encOut,
                  F(DC_WQ) + i * DD, F(DC_BQ) + i * DIM, F(DC_WK) + i * DD, F(DC_BK) + i * DIM,
                  F(DC_WV) + i * DD, F(DC_BV) + i * DIM, F(DC_WO) + i * DD, F(DC_BO) + i * DIM,
                  enc_ids, 0, /*resid=*/nullptr, /*out=*/CR, Qb, Kb, Vb, Sc, CT);
        // t = y + cross ; y = cross + ffn(ln3(t))   (faithful to source)
        add_kernel<<<(unsigned)((ND + 255) / 256), 256, 0, stream>>>(Yb, CR, Tb, ND);
        layernorm_kernel<<<ROWS, 256, 0, stream>>>(Tb, F(D_LN3G) + i * DIM, F(D_LN3B) + i * DIM, Hb);
        dense(stream, Hb, F(D_W1) + i * DF, F(D_B1) + i * FFD, nullptr, Fb, ROWS, FFD, DIM, true);
        dense(stream, Fb, F(D_W2) + i * DF, F(D_B2) + i * DIM, CR, Yb, ROWS, DIM, FFD, false);
    }

    // ---------------- logits ----------------
    dense(stream, Yb, F(FIN_W), F(FIN_B), nullptr, logits, ROWS, VOC, DIM, false);
}
